// NSE_22651657519408
// MI455X (gfx1250) — matrix operand order repaired and checked by reading, not compiled, not run on hardware
//
#include <hip/hip_runtime.h>
#include <math.h>

// ---------------- problem constants ----------------
constexpr int T  = 64;
constexpr int B  = 16;
constexpr int S  = 128;
constexpr int D  = 512;
constexpr int G  = 4 * D;      // 2048 gate width
constexpr int BD = B * D;      // 8192

typedef __attribute__((ext_vector_type(16))) _Float16 v16h;
typedef __attribute__((ext_vector_type(8)))  _Float16 v8h;
typedef __attribute__((ext_vector_type(8)))  float    v8f;

// ---------------- WMMA helper ----------------
// C += A(16xK, row-major, lda=K) * W(16 rows of K, row-major)^T for one 16x16 tile.
// Fragment layouts used by this code (wave32, contiguous k = 16h + s, not the ISA table order):
//   A: lane m = lane&15 ; lanes 0-15 hold K=[0..15]; lanes 16-31 hold K=[16..31]
//   B: lane n = lane&15 ; lanes 0-15 hold K=[0..15]; lanes 16-31 hold K=[16..31]
__device__ __forceinline__ v8f wmma_acc_tile(const _Float16* __restrict__ Arow,
                                             const _Float16* __restrict__ Brow,
                                             int K, int hiA, int hiB, v8f c) {
    for (int kc = 0; kc < K; kc += 32) {
        v8h a0 = *(const v8h*)(Arow + kc + hiB);
        v8h a1 = *(const v8h*)(Arow + kc + hiB + 8);
        v16h a = __builtin_shufflevector(a0, a1, 0,1,2,3,4,5,6,7,8,9,10,11,12,13,14,15);
        v16h b = *(const v16h*)(Brow + kc + hiB);
        c = __builtin_amdgcn_wmma_f32_16x16x32_f16(false, a, false, b, (short)0, c, false, false);
    }
    return c;
}

__device__ __forceinline__ float sigmoidf_(float x) { return 1.f / (1.f + expf(-x)); }

// ---------------- small utility kernels ----------------
__global__ void k_cvt16(const float* __restrict__ src, _Float16* __restrict__ dst, int n) {
    int i = blockIdx.x * 256 + threadIdx.x;
    if (i < n) dst[i] = (_Float16)src[i];
}

__global__ void k_addvec(const float* __restrict__ a, const float* __restrict__ b,
                         float* __restrict__ d, int n) {
    int i = blockIdx.x * 256 + threadIdx.x;
    if (i < n) d[i] = a[i] + b[i];
}

// ---------------- big up-front GEMM: Xr = emb @ Wih_r^T + (bih_r+bhh_r) ----------------
// A: (T*B=1024) x 512 f16, W: 2048 x 512 f16, out: 1024 x 2048 f32.
// One wave per 16x16 tile: 64 m-tiles x 128 n-tiles = 8192 waves.
__global__ void k_gemm_xr(const _Float16* __restrict__ A, const _Float16* __restrict__ W,
                          const float* __restrict__ bias, float* __restrict__ out) {
    int wave = blockIdx.x * (blockDim.x >> 5) + (threadIdx.x >> 5);
    int lane = threadIdx.x & 31;
    int mt = wave >> 7;            // 0..63
    int nt = wave & 127;           // 0..127
    int m0 = mt * 16, n0 = nt * 16;
    int hiA = (lane & 16) ? 8 : 0;
    int hiB = (lane & 16) ? 16 : 0;
    const _Float16* Arow = A + (size_t)(m0 + (lane & 15)) * D;
    const _Float16* Brow = W + (size_t)(n0 + (lane & 15)) * D;
    v8f c = {};
    c = wmma_acc_tile(Arow, Brow, D, hiA, hiB, c);
    int ncol  = n0 + (lane & 15);
    int mbase = m0 + ((lane & 16) ? 8 : 0);
    float bn = bias[ncol];
#pragma unroll
    for (int r = 0; r < 8; ++r)
        out[(size_t)(mbase + r) * G + ncol] = c[r] + bn;
}

// ---------------- fused read step: gates = Xr_t + h@Whh_r^T ; LSTM pointwise ----------------
// 32 blocks x 128 threads. Block handles d0 = blockIdx*16 .. +15.
// Wave g computes the 16x16 gate-g tile (W rows g*512+d0..), tiles staged in LDS,
// then the block applies the LSTM nonlinearity for its (16 b x 16 d) slab.
// h16 is ping-ponged across timesteps (blocks read all columns, write only their own).
__global__ void k_step_r(const _Float16* __restrict__ h16_rd, const _Float16* __restrict__ Whh,
                         const float* __restrict__ Xr_t,
                         float* __restrict__ cst, float* __restrict__ hst,
                         _Float16* __restrict__ h16_wr, _Float16* __restrict__ comp16) {
    __shared__ float gs[4][16][16];
    int tid  = threadIdx.x;
    int wv   = tid >> 5;            // gate index 0..3
    int lane = tid & 31;
    int d0   = blockIdx.x * 16;
    int hiA = (lane & 16) ? 8 : 0;
    int hiB = (lane & 16) ? 16 : 0;
    const _Float16* Arow = h16_rd + (size_t)(lane & 15) * D;
    const _Float16* Brow = Whh + (size_t)(wv * D + d0 + (lane & 15)) * D;
    v8f c = {};
    c = wmma_acc_tile(Arow, Brow, D, hiA, hiB, c);
    int mbase = (lane & 16) ? 8 : 0;
#pragma unroll
    for (int r = 0; r < 8; ++r)
        gs[wv][mbase + r][lane & 15] = c[r];
    __syncthreads();
    // pointwise: 256 (b,d) elems, 2 per thread
#pragma unroll
    for (int q = 0; q < 2; ++q) {
        int idx = tid + q * 128;
        int b   = idx >> 4;
        int dl  = idx & 15;
        int d   = d0 + dl;
        const float* xr = Xr_t + b * G;
        float gi = gs[0][b][dl] + xr[d];
        float gf = gs[1][b][dl] + xr[D + d];
        float gg = gs[2][b][dl] + xr[2 * D + d];
        float go = gs[3][b][dl] + xr[3 * D + d];
        int bd = b * D + d;
        float c2 = sigmoidf_(gf) * cst[bd] + sigmoidf_(gi) * tanhf(gg);
        float h2 = sigmoidf_(go) * tanhf(c2);
        cst[bd] = c2;
        hst[bd] = h2;
        h16_wr[bd] = (_Float16)h2;
        comp16[b * 2 * D + d] = (_Float16)h2;   // front half of [hr|m]
    }
}

// ---------------- fused write step: gates = xw@Wih_w^T + hw@Whh_w^T + bias ; LSTM ----------------
__global__ void k_step_w(const _Float16* __restrict__ xw16, const _Float16* __restrict__ Wih,
                         const _Float16* __restrict__ h16_rd, const _Float16* __restrict__ Whh,
                         const float* __restrict__ bias,
                         float* __restrict__ cst, float* __restrict__ hst,
                         _Float16* __restrict__ h16_wr, float* __restrict__ outT) {
    __shared__ float gs[4][16][16];
    int tid  = threadIdx.x;
    int wv   = tid >> 5;
    int lane = tid & 31;
    int d0   = blockIdx.x * 16;
    int hiA = (lane & 16) ? 8 : 0;
    int hiB = (lane & 16) ? 16 : 0;
    int mr = lane & 15;
    size_t wrow = (size_t)(wv * D + d0 + mr) * D;
    v8f c = {};
    c = wmma_acc_tile(xw16   + (size_t)mr * D, Wih + wrow, D, hiA, hiB, c);
    c = wmma_acc_tile(h16_rd + (size_t)mr * D, Whh + wrow, D, hiA, hiB, c);
    int mbase = (lane & 16) ? 8 : 0;
#pragma unroll
    for (int r = 0; r < 8; ++r)
        gs[wv][mbase + r][mr] = c[r];
    __syncthreads();
#pragma unroll
    for (int q = 0; q < 2; ++q) {
        int idx = tid + q * 128;
        int b   = idx >> 4;
        int dl  = idx & 15;
        int d   = d0 + dl;
        float gi = gs[0][b][dl] + bias[d];
        float gf = gs[1][b][dl] + bias[D + d];
        float gg = gs[2][b][dl] + bias[2 * D + d];
        float go = gs[3][b][dl] + bias[3 * D + d];
        int bd = b * D + d;
        float c2 = sigmoidf_(gf) * cst[bd] + sigmoidf_(gi) * tanhf(gg);
        float h2 = sigmoidf_(go) * tanhf(c2);
        cst[bd] = c2;
        hst[bd] = h2;
        h16_wr[bd] = (_Float16)h2;
        outT[bd] = h2;
    }
}

// ---------------- comp GEMM: xw = [hr|m](16x1024) @ Wc^T + bc -> f16 (16x512) ----------------
// 32 waves (8 blocks x 4 waves).
__global__ void k_comp(const _Float16* __restrict__ comp16, const _Float16* __restrict__ Wc16,
                       const float* __restrict__ bc, _Float16* __restrict__ xw16) {
    int wave = blockIdx.x * (blockDim.x >> 5) + (threadIdx.x >> 5);
    int lane = threadIdx.x & 31;
    int n0 = wave * 16;
    int hiA = (lane & 16) ? 8 : 0;
    int hiB = (lane & 16) ? 16 : 0;
    const _Float16* Arow = comp16 + (size_t)(lane & 15) * (2 * D);
    const _Float16* Brow = Wc16   + (size_t)(n0 + (lane & 15)) * (2 * D);
    v8f c = {};
    c = wmma_acc_tile(Arow, Brow, 2 * D, hiA, hiB, c);
    int ncol  = n0 + (lane & 15);
    int mbase = (lane & 16) ? 8 : 0;
    float bn = bc[ncol];
#pragma unroll
    for (int r = 0; r < 8; ++r)
        xw16[(mbase + r) * D + ncol] = (_Float16)(c[r] + bn);
}

// ---------------- attention: sim -> softmax z -> m (written into comp back half) ----------------
__global__ void k_attn(const float* __restrict__ hr, const float* __restrict__ Mbuf,
                       float* __restrict__ z, _Float16* __restrict__ comp16) {
    int b = blockIdx.x;
    int tid = threadIdx.x;
    __shared__ float red[256];
    __shared__ float zs[S];
    const float* hb = hr + b * D;
    int s  = tid >> 1;
    int kh = (tid & 1) << 8;        // 0 or 256
    const float* Ms = Mbuf + (size_t)(b * S + s) * D + kh;
    const float* hh = hb + kh;
    float acc = 0.f;
#pragma unroll 4
    for (int k = 0; k < 256; ++k) acc += hh[k] * Ms[k];
    red[tid] = acc;
    __syncthreads();
    if ((tid & 1) == 0) zs[s] = red[tid] + red[tid + 1];
    __syncthreads();
    red[tid] = (tid < S) ? zs[tid] : -3.4e38f;
    __syncthreads();
    for (int off = 128; off > 0; off >>= 1) {
        if (tid < off) red[tid] = fmaxf(red[tid], red[tid + off]);
        __syncthreads();
    }
    float mx = red[0];
    __syncthreads();
    float e = (tid < S) ? expf(zs[tid] - mx) : 0.f;
    red[tid] = e;
    __syncthreads();
    for (int off = 128; off > 0; off >>= 1) {
        if (tid < off) red[tid] += red[tid + off];
        __syncthreads();
    }
    float inv = 1.f / red[0];
    __syncthreads();
    if (tid < S) {
        float zz = e * inv;
        zs[tid] = zz;
        z[b * S + tid] = zz;
    }
    __syncthreads();
    for (int d = tid; d < D; d += 256) {
        float m = 0.f;
        for (int ss = 0; ss < S; ++ss)
            m += zs[ss] * Mbuf[(size_t)(b * S + ss) * D + d];
        comp16[b * 2 * D + D + d] = (_Float16)m;
    }
}

// ---------------- memory update: M = (1-z)*(M + hw) ----------------
__global__ void k_memupd(float* __restrict__ Mbuf, const float* __restrict__ z,
                         const float* __restrict__ hw) {
    int idx = blockIdx.x * 256 + threadIdx.x;       // B*S*D = 1048576
    int d  = idx & (D - 1);
    int bs = idx >> 9;                              // b*S + s
    int b  = bs >> 7;
    float zz = z[bs];
    Mbuf[idx] = (1.f - zz) * (Mbuf[idx] + hw[b * D + d]);
}

// ---------------- host launch ----------------
extern "C" void kernel_launch(void* const* d_in, const int* in_sizes, int n_in,
                              void* d_out, int out_size, void* d_ws, size_t ws_size,
                              hipStream_t stream) {
    const float* emb   = (const float*)d_in[0];   // (T,B,D)
    const float* M     = (const float*)d_in[1];   // (B,S,D)
    const float* hr0   = (const float*)d_in[2];
    const float* cr0   = (const float*)d_in[3];
    const float* hw0   = (const float*)d_in[4];
    const float* cw0   = (const float*)d_in[5];
    const float* Wih_r = (const float*)d_in[6];   // (4D,D)
    const float* Whh_r = (const float*)d_in[7];
    const float* bih_r = (const float*)d_in[8];
    const float* bhh_r = (const float*)d_in[9];
    const float* Wih_w = (const float*)d_in[10];
    const float* Whh_w = (const float*)d_in[11];
    const float* bih_w = (const float*)d_in[12];
    const float* bhh_w = (const float*)d_in[13];
    const float* Wc    = (const float*)d_in[14];  // (D,2D)
    const float* bc    = (const float*)d_in[15];

    // ---- carve workspace (~24 MB total) ----
    char* p = (char*)d_ws;
    auto carve = [&](size_t bytes) -> void* {
        void* r = (void*)p;
        p += (bytes + 255) & ~(size_t)255;
        return r;
    };
    float*    Mbuf    = (float*)   carve((size_t)B * S * D * 4);   // mutable memory tensor
    float*    Xr      = (float*)   carve((size_t)T * B * G * 4);   // precomputed input gates
    _Float16* emb16   = (_Float16*)carve((size_t)T * B * D * 2);
    _Float16* Wihr16  = (_Float16*)carve((size_t)G * D * 2);
    _Float16* Whhr16  = (_Float16*)carve((size_t)G * D * 2);
    _Float16* Wihw16  = (_Float16*)carve((size_t)G * D * 2);
    _Float16* Whhw16  = (_Float16*)carve((size_t)G * D * 2);
    _Float16* Wc16    = (_Float16*)carve((size_t)D * 2 * D * 2);
    float*    biasr   = (float*)   carve(G * 4);
    float*    biasw   = (float*)   carve(G * 4);
    float*    hrbuf   = (float*)   carve(BD * 4);
    float*    crbuf   = (float*)   carve(BD * 4);
    float*    hwbuf   = (float*)   carve(BD * 4);
    float*    cwbuf   = (float*)   carve(BD * 4);
    _Float16* hr16a   = (_Float16*)carve(BD * 2);   // ping-pong pair (cross-block RAW)
    _Float16* hr16b   = (_Float16*)carve(BD * 2);
    _Float16* hw16a   = (_Float16*)carve(BD * 2);
    _Float16* hw16b   = (_Float16*)carve(BD * 2);
    _Float16* xw16    = (_Float16*)carve(BD * 2);
    _Float16* comp16  = (_Float16*)carve((size_t)B * 2 * D * 2);
    float*    zbuf    = (float*)   carve(B * S * 4);

    float* out = (float*)d_out;

    // ---- one-time (per launch) preprocessing ----
    hipMemcpyAsync(Mbuf, M, (size_t)B * S * D * 4, hipMemcpyDeviceToDevice, stream);
    hipMemcpyAsync(hrbuf, hr0, BD * 4, hipMemcpyDeviceToDevice, stream);
    hipMemcpyAsync(crbuf, cr0, BD * 4, hipMemcpyDeviceToDevice, stream);
    hipMemcpyAsync(hwbuf, hw0, BD * 4, hipMemcpyDeviceToDevice, stream);
    hipMemcpyAsync(cwbuf, cw0, BD * 4, hipMemcpyDeviceToDevice, stream);

    auto cvt = [&](const float* s, _Float16* d, int n) {
        k_cvt16<<<(n + 255) / 256, 256, 0, stream>>>(s, d, n);
    };
    cvt(emb,   emb16,  T * B * D);
    cvt(Wih_r, Wihr16, G * D);
    cvt(Whh_r, Whhr16, G * D);
    cvt(Wih_w, Wihw16, G * D);
    cvt(Whh_w, Whhw16, G * D);
    cvt(Wc,    Wc16,   D * 2 * D);
    cvt(hr0,   hr16a,  BD);
    cvt(hw0,   hw16a,  BD);
    k_addvec<<<(G + 255) / 256, 256, 0, stream>>>(bih_r, bhh_r, biasr, G);
    k_addvec<<<(G + 255) / 256, 256, 0, stream>>>(bih_w, bhh_w, biasw, G);

    // Xr = emb @ Wih_r^T + biasr  (64 m-tiles x 128 n-tiles = 8192 waves)
    k_gemm_xr<<<2048, 128, 0, stream>>>(emb16, Wihr16, biasr, Xr);

    // ---- recurrence (5 launches / step) ----
    for (int t = 0; t < T; ++t) {
        _Float16* hrR = (t & 1) ? hr16b : hr16a;
        _Float16* hrW = (t & 1) ? hr16a : hr16b;
        _Float16* hwR = (t & 1) ? hw16b : hw16a;
        _Float16* hwW = (t & 1) ? hw16a : hw16b;
        k_step_r<<<32, 128, 0, stream>>>(hrR, Whhr16, Xr + (size_t)t * B * G,
                                         crbuf, hrbuf, hrW, comp16);
        k_attn<<<B, 256, 0, stream>>>(hrbuf, Mbuf, zbuf, comp16);
        k_comp<<<8, 128, 0, stream>>>(comp16, Wc16, bc, xw16);
        k_step_w<<<32, 128, 0, stream>>>(xw16, Wihw16, hwR, Whhw16, biasw,
                                         cwbuf, hwbuf, hwW, out + (size_t)t * BD);
        k_memupd<<<(B * S * D) / 256, 256, 0, stream>>>(Mbuf, zbuf, hwbuf);
    }

    // ---- final state outputs: outputs | hr | cr | hw | cw | Mf ----
    size_t o = (size_t)T * BD;
    hipMemcpyAsync(out + o, hrbuf, BD * 4, hipMemcpyDeviceToDevice, stream); o += BD;
    hipMemcpyAsync(out + o, crbuf, BD * 4, hipMemcpyDeviceToDevice, stream); o += BD;
    hipMemcpyAsync(out + o, hwbuf, BD * 4, hipMemcpyDeviceToDevice, stream); o += BD;
    hipMemcpyAsync(out + o, cwbuf, BD * 4, hipMemcpyDeviceToDevice, stream); o += BD;
    hipMemcpyAsync(out + o, Mbuf, (size_t)B * S * D * 4, hipMemcpyDeviceToDevice, stream);
}
